// GraphSAGE_20469814133412
// MI455X (gfx1250) — matrix operand order repaired and checked by reading, not compiled, not run on hardware
//
#include <hip/hip_runtime.h>
#include <hip/hip_bf16.h>

typedef __attribute__((ext_vector_type(16))) _Float16 v16h;
typedef __attribute__((ext_vector_type(8)))  float    v8f;

#define N_NODES 50000
#define N_EDGES 800000
#define IN_DIM  128
#define HID_DIM 256
#define OUT_DIM 40
#define OUT_PAD 48

static inline int ceil_div(long a, int b) { return (int)((a + b - 1) / b); }

// ---------------------------------------------------------------- utilities
__global__ __launch_bounds__(256) void zero_f32(float* __restrict__ p, long n) {
  long i = (long)blockIdx.x * 256 + threadIdx.x;
  if (i < n) p[i] = 0.f;
}

// Pack f32 weights [K x N] row-major into f16 WMMA B-fragment layout.
// Fragment element (nt, kstep, lane, i) = W[kstep*32 + (lane>>4)*16 + i][nt*16 + (lane&15)]
__global__ __launch_bounds__(256) void pack_w(const float* __restrict__ W,
                                              _Float16* __restrict__ Wp,
                                              int K, int N, int Npad) {
  int Kt = K >> 5;
  int total = (Npad >> 4) * Kt * 512;
  int idx = blockIdx.x * 256 + threadIdx.x;
  if (idx >= total) return;
  int i    = idx & 15;
  int lane = (idx >> 4) & 31;
  int kstep = (idx >> 9) % Kt;
  int nt    = idx / (Kt << 9);
  int n = nt * 16 + (lane & 15);
  int k = kstep * 32 + (lane >> 4) * 16 + i;   // B layout: lane-half holds 16 consecutive K
  float v = (n < N) ? W[(long)k * N + n] : 0.f;
  Wp[idx] = (_Float16)v;
}

// ---------------------------------------------------------------- aggregation
__global__ __launch_bounds__(256) void edge_count(const int* __restrict__ ei,
                                                  float* __restrict__ cnt) {
  int e = blockIdx.x * 256 + threadIdx.x;
  if (e < N_EDGES) atomicAdd(&cnt[ei[N_EDGES + e]], 1.f);
}

// One thread per (edge, 4-dim chunk): float4 gather from X[src], 4 f32 atomics into agg[dst]
__global__ __launch_bounds__(256) void edge_scatter(const int* __restrict__ ei,
                                                    const float* __restrict__ X,
                                                    float* __restrict__ agg,
                                                    int D, int qshift) {
  long t = (long)blockIdx.x * 256 + threadIdx.x;
  long total = (long)N_EDGES << qshift;
  if (t >= total) return;
  int e = (int)(t >> qshift);
  int c = (int)(t & ((1 << qshift) - 1)) << 2;
  int s = ei[e];
  int d = ei[N_EDGES + e];
  const float4 v = *(const float4*)(X + (long)s * D + c);
  float* o = agg + (long)d * D + c;
  atomicAdd(o + 0, v.x);
  atomicAdd(o + 1, v.y);
  atomicAdd(o + 2, v.z);
  atomicAdd(o + 3, v.w);
}

__global__ __launch_bounds__(256) void mean_div(float* __restrict__ agg,
                                                const float* __restrict__ cnt,
                                                int D) {
  long t = (long)blockIdx.x * 256 + threadIdx.x;
  if (t >= (long)N_NODES * D) return;
  float c = cnt[t / D];
  agg[t] *= (c > 1.f) ? (1.f / c) : 1.f;
}

// ---------------------------------------------------------------- dual-WMMA GEMM
// C[M x Nout] = act( A1[M x K] @ B1 + A2[M x K] @ B2 + bias ),  B* pre-packed f16 fragments
// One 16-row M-tile per block (256 thr = 8 waves); wave w owns N-tiles w, w+8, ...
__global__ __launch_bounds__(256) void gemm_dual_wmma(
    const float* __restrict__ A1, const float* __restrict__ A2,
    const _Float16* __restrict__ B1p, const _Float16* __restrict__ B2p,
    const float* __restrict__ bias, float* __restrict__ C,
    int M, int K, int Npad, int Nout, int do_relu) {
  __shared__ __attribute__((aligned(32))) _Float16 ldsA1[8 * 512];  // up to K=256
  __shared__ __attribute__((aligned(32))) _Float16 ldsA2[8 * 512];

  const int tid  = threadIdx.x;
  const int lane = tid & 31;
  const int wave = tid >> 5;
  const int Kt   = K >> 5;            // K-steps of 32
  const int row0 = blockIdx.x * 16;

  // Stage both A tiles (f32 -> f16) in WMMA A-fragment layout:
  // slot(kstep, l, i): row = l&15, hs = l>>4, K = kstep*32 + hs*16 + i
  for (int idx = tid; idx < 2 * Kt * 512; idx += 256) {
    int sel = (idx >= Kt * 512);
    int j = sel ? idx - Kt * 512 : idx;
    int i = j & 15;
    int l = (j >> 4) & 31;
    int kstep = j >> 9;
    int m  = l & 15;
    int hs = l >> 4;
    int koff = hs * 16 + i;
    int k = kstep * 32 + koff;
    int row = row0 + m;
    if (row >= M) row = M - 1;        // clamp (harmless: store is guarded)
    const float* A = sel ? A2 : A1;
    _Float16 v = (_Float16)A[(long)row * K + k];
    if (sel) ldsA2[j] = v; else ldsA1[j] = v;
  }
  __syncthreads();

  const int ntiles = Npad >> 4;
  for (int nt = wave; nt < ntiles; nt += 8) {   // wave-uniform loop (EXEC all-ones for WMMA)
    v8f acc = {};
    const _Float16* bp1 = B1p + (long)nt * Kt * 512 + lane * 16;
    const _Float16* bp2 = B2p + (long)nt * Kt * 512 + lane * 16;
    for (int ks = 0; ks < Kt; ++ks) {
      v16h a = *(const v16h*)&ldsA1[ks * 512 + lane * 16];
      v16h b = *(const v16h*)(bp1 + ks * 512);
      acc = __builtin_amdgcn_wmma_f32_16x16x32_f16(false, a, false, b, (short)0, acc,
                                                   false, false);
    }
    for (int ks = 0; ks < Kt; ++ks) {
      v16h a = *(const v16h*)&ldsA2[ks * 512 + lane * 16];
      v16h b = *(const v16h*)(bp2 + ks * 512);
      acc = __builtin_amdgcn_wmma_f32_16x16x32_f16(false, a, false, b, (short)0, acc,
                                                   false, false);
    }
    // C layout: lane l -> n = l&15, rows r + 8*(l>>4), r = 0..7
    int n = nt * 16 + (lane & 15);
    if (n < Nout) {
      float bv = bias[n];
      int rbase = row0 + ((lane >> 4) << 3);
      #pragma unroll
      for (int r = 0; r < 8; ++r) {
        int row = rbase + r;
        if (row < M) {
          float v = acc[r] + bv;
          if (do_relu) v = (v > 0.f) ? v : 0.f;
          C[(long)row * Nout + n] = v;
        }
      }
    }
  }
}

// ---------------------------------------------------------------- log-softmax (40 classes)
__global__ __launch_bounds__(256) void logsoftmax40(const float* __restrict__ logits,
                                                    float* __restrict__ out) {
  int nrow = blockIdx.x * 256 + threadIdx.x;
  if (nrow >= N_NODES) return;
  const float* p = logits + (long)nrow * OUT_DIM;
  float v[OUT_DIM];
  float mx = -1e30f;
  #pragma unroll
  for (int i = 0; i < OUT_DIM; ++i) { v[i] = p[i]; mx = fmaxf(mx, v[i]); }
  float s = 0.f;
  #pragma unroll
  for (int i = 0; i < OUT_DIM; ++i) s += __expf(v[i] - mx);
  float lse = mx + __logf(s);
  float* o = out + (long)nrow * OUT_DIM;
  #pragma unroll
  for (int i = 0; i < OUT_DIM; ++i) o[i] = v[i] - lse;
}

// ---------------------------------------------------------------- launch
extern "C" void kernel_launch(void* const* d_in, const int* in_sizes, int n_in,
                              void* d_out, int out_size, void* d_ws, size_t ws_size,
                              hipStream_t stream) {
  const float* x   = (const float*)d_in[0];
  const int*   ei  = (const int*)d_in[1];   // [2, E]: row 0 = src, row 1 = dst
  const float* W1l = (const float*)d_in[2];
  const float* W1r = (const float*)d_in[3];
  const float* b1  = (const float*)d_in[4];
  const float* W2l = (const float*)d_in[5];
  const float* W2r = (const float*)d_in[6];
  const float* b2  = (const float*)d_in[7];
  float* out = (float*)d_out;

  char* ws = (char*)d_ws;
  size_t off = 0;
  auto carve = [&](size_t bytes) -> void* {
    void* p = ws + off;
    off += (bytes + 31) & ~(size_t)31;
    return p;
  };
  float* agg1   = (float*)carve((size_t)N_NODES * IN_DIM  * 4);  // 25.6 MB
  float* cnt    = (float*)carve((size_t)N_NODES * 4);
  float* h      = (float*)carve((size_t)N_NODES * HID_DIM * 4);  // 51.2 MB
  float* agg2   = (float*)carve((size_t)N_NODES * HID_DIM * 4);  // 51.2 MB
  float* logits = (float*)carve((size_t)N_NODES * OUT_DIM * 4);  // 8 MB
  _Float16* W1lp = (_Float16*)carve((HID_DIM / 16) * (IN_DIM  / 32) * 512 * 2);
  _Float16* W1rp = (_Float16*)carve((HID_DIM / 16) * (IN_DIM  / 32) * 512 * 2);
  _Float16* W2lp = (_Float16*)carve((OUT_PAD / 16) * (HID_DIM / 32) * 512 * 2);
  _Float16* W2rp = (_Float16*)carve((OUT_PAD / 16) * (HID_DIM / 32) * 512 * 2);

  // Pack weights into WMMA B-fragment f16 layout
  {
    int t1 = (HID_DIM / 16) * (IN_DIM / 32) * 512;
    int t2 = (OUT_PAD / 16) * (HID_DIM / 32) * 512;
    pack_w<<<ceil_div(t1, 256), 256, 0, stream>>>(W1l, W1lp, IN_DIM,  HID_DIM, HID_DIM);
    pack_w<<<ceil_div(t1, 256), 256, 0, stream>>>(W1r, W1rp, IN_DIM,  HID_DIM, HID_DIM);
    pack_w<<<ceil_div(t2, 256), 256, 0, stream>>>(W2l, W2lp, HID_DIM, OUT_DIM, OUT_PAD);
    pack_w<<<ceil_div(t2, 256), 256, 0, stream>>>(W2r, W2rp, HID_DIM, OUT_DIM, OUT_PAD);
  }

  // ---- Layer 1: mean aggregation of x, then h = relu(agg1@W1l + x@W1r + b1)
  zero_f32<<<ceil_div((long)N_NODES * IN_DIM, 256), 256, 0, stream>>>(agg1, (long)N_NODES * IN_DIM);
  zero_f32<<<ceil_div((long)N_NODES, 256), 256, 0, stream>>>(cnt, (long)N_NODES);
  edge_count<<<ceil_div((long)N_EDGES, 256), 256, 0, stream>>>(ei, cnt);
  edge_scatter<<<ceil_div((long)N_EDGES * (IN_DIM / 4), 256), 256, 0, stream>>>(
      ei, x, agg1, IN_DIM, 5);
  mean_div<<<ceil_div((long)N_NODES * IN_DIM, 256), 256, 0, stream>>>(agg1, cnt, IN_DIM);
  gemm_dual_wmma<<<N_NODES / 16, 256, 0, stream>>>(
      agg1, x, W1lp, W1rp, b1, h, N_NODES, IN_DIM, HID_DIM, HID_DIM, 1);

  // ---- Layer 2: mean aggregation of h, then logits = agg2@W2l + h@W2r + b2
  zero_f32<<<ceil_div((long)N_NODES * HID_DIM, 256), 256, 0, stream>>>(agg2, (long)N_NODES * HID_DIM);
  edge_scatter<<<ceil_div((long)N_EDGES * (HID_DIM / 4), 256), 256, 0, stream>>>(
      ei, h, agg2, HID_DIM, 6);
  mean_div<<<ceil_div((long)N_NODES * HID_DIM, 256), 256, 0, stream>>>(agg2, cnt, HID_DIM);
  gemm_dual_wmma<<<N_NODES / 16, 256, 0, stream>>>(
      agg2, h, W2lp, W2rp, b2, logits, N_NODES, HID_DIM, OUT_PAD, OUT_DIM, 0);

  // ---- log_softmax over 40 classes
  logsoftmax40<<<ceil_div((long)N_NODES, 256), 256, 0, stream>>>(logits, out);
}
